// HGT_35811437314878
// MI455X (gfx1250) — matrix operand order repaired and checked by reading, not compiled, not run on hardware
//
#include <hip/hip_runtime.h>
#include <hip/hip_bf16.h>
#include <math.h>

typedef _Float16 v16h __attribute__((ext_vector_type(16)));
typedef _Float16 v8h  __attribute__((ext_vector_type(8)));
typedef _Float16 v4h  __attribute__((ext_vector_type(4)));
typedef float    v8f  __attribute__((ext_vector_type(8)));

namespace {
constexpr int kNB  = 190000;
constexpr int kNG  = 60000;
constexpr int kNBp = 190016;   // padded to 64-row GEMM tiles
constexpr int kNGp = 60032;
constexpr int kHID = 128;
constexpr int kNH  = 40000;    // 10000 graphs * 4 readout nodes
constexpr int kEBB = 1200000, kEGB = 300000, kEBG = 300000;
constexpr int kPerGraph = 19;
constexpr float kInvSqrtD = 0.25f;   // 1/sqrt(16)
}

// ---------------------------------------------------------------- input MLP
__global__ __launch_bounds__(256) void k_in_linear_relu(
    const float* __restrict__ x, const float* __restrict__ W,
    const float* __restrict__ b, float* __restrict__ y, int n) {
  int i = blockIdx.x * 256 + threadIdx.x;
  if (i >= n * kHID) return;
  int row = i >> 7, c = i & 127;
  const float* xr = x + (size_t)row * 8;
  float acc = b[c];
#pragma unroll
  for (int f = 0; f < 8; ++f) acc = fmaf(xr[f], W[f * kHID + c], acc);
  y[i] = fmaxf(acc, 0.f);
}

// ------------------------------------------------- WMMA GEMM  Y = act(X@W+b)
// X:[N,128] f32, W:[128,128] f32, Y:[N,128] f32. N multiple of 64 (padded).
// 256 threads = 8 waves; wave w owns column tile w (16 cols), 4 row tiles.
// W is staged into LDS pre-swizzled into per-(wave,kstep,lane) fragment
// order so each B operand is two ds_load_b128 with no VALU repack.
__global__ __launch_bounds__(256) void k_gemm128_wmma(
    const float* __restrict__ X, const float* __restrict__ W,
    const float* __restrict__ bias, float* __restrict__ Y, int act) {
  __shared__ _Float16 As[64][kHID + 8];          // 17408 B (+8: bank spread)
  __shared__ _Float16 Bf[8 * 4 * 32 * 16];       // 32768 B, fragment order
  const int t = threadIdx.x;
  const size_t row0 = (size_t)blockIdx.x * 64;

  // ---- stage W: coalesced float4 reads, swizzled 16-bit LDS writes.
  // fragment(w,s,lane=hs*16+ml,e) = W[k=s*32+hs*16+e][n=w*16+ml]
  for (int i = t * 4; i < kHID * kHID; i += 256 * 4) {
    const float4 w4 = *(const float4*)(W + i);
    const int k = i >> 7;
    const int s = k >> 5, hsv = (k >> 4) & 1, e = k & 15;
    const float wv4[4] = {w4.x, w4.y, w4.z, w4.w};
#pragma unroll
    for (int j = 0; j < 4; ++j) {
      const int n = (i & 127) + j;
      const int wv = n >> 4, ml = n & 15;
      Bf[(((wv * 4 + s) * 32) + hsv * 16 + ml) * 16 + e] = (_Float16)wv4[j];
    }
  }
  // ---- stage X tile: float4 global reads -> 4 contiguous f16 LDS stores
  for (int i = t * 4; i < 64 * kHID; i += 256 * 4) {
    const float4 x4 = *(const float4*)(X + (row0 << 7) + i);
    const int r = i >> 7, c = i & 127;
    v4h h4;
    h4[0] = (_Float16)x4.x; h4[1] = (_Float16)x4.y;
    h4[2] = (_Float16)x4.z; h4[3] = (_Float16)x4.w;
    *(v4h*)&As[r][c] = h4;
  }
  __syncthreads();

  const int lane = t & 31, wave = t >> 5;
  const int ml = lane & 15, hs = lane >> 4;
  const int n0 = wave * 16;

  // B fragments: contiguous 32 B per (wave, kstep, lane)
  v16h bf[4];
#pragma unroll
  for (int s = 0; s < 4; ++s)
    bf[s] = *(const v16h*)&Bf[(((wave * 4 + s) * 32) + lane) * 16];

  const float bc = bias ? bias[n0 + ml] : 0.f;

#pragma unroll
  for (int rt = 0; rt < 4; ++rt) {
    v8f c = {};
#pragma unroll
    for (int s = 0; s < 4; ++s) {
      // A fragment (16x32): lanes 0-15: K 0-15 ; lanes 16-31: K 16-31
      const v8h alo = *(const v8h*)&As[rt * 16 + ml][s * 32 + hs * 16];
      const v8h ahi = *(const v8h*)&As[rt * 16 + ml][s * 32 + hs * 16 + 8];
      const v16h a = __builtin_shufflevector(alo, ahi, 0, 1, 2, 3, 4, 5, 6, 7,
                                             8, 9, 10, 11, 12, 13, 14, 15);
      c = __builtin_amdgcn_wmma_f32_16x16x32_f16(false, a, false, bf[s],
                                                 (short)0, c, false, false);
    }
#pragma unroll
    for (int j = 0; j < 8; ++j) {   // D: VGPR j -> row j + 8*hs, col = ml
      float v = c[j] + bc;
      if (act == 1) v = fmaxf(v, 0.f);
      Y[(row0 + rt * 16 + j + 8 * hs) * kHID + n0 + ml] = v;
    }
  }
}

// ------------------------------------ per-relation block-diagonal 128x128 W
// Wblk[h*16+d][h*16+e] = rel[h][d][e]  (zero off the 8 head blocks)
__global__ __launch_bounds__(256) void k_blockdiag(
    const float* __restrict__ rel, float* __restrict__ Wblk) {
  int i = blockIdx.x * 256 + threadIdx.x;   // exactly 16384
  int row = i >> 7, col = i & 127;
  int hr = row >> 4, hc = col >> 4;
  Wblk[i] = (hr == hc) ? rel[(hr * 16 + (row & 15)) * 16 + (col & 15)] : 0.f;
}

// -------------------------------------------------------- edge attention
__device__ __forceinline__ unsigned f2mono(float f) {
  unsigned u = __float_as_uint(f);
  return (u & 0x80000000u) ? ~u : (u | 0x80000000u);
}
__device__ __forceinline__ float mono2f(unsigned u) {
  return __uint_as_float((u & 0x80000000u) ? (u ^ 0x80000000u) : ~u);
}

__global__ __launch_bounds__(256) void k_edge_alpha(
    const int* __restrict__ src, const int* __restrict__ dst,
    const float* __restrict__ kt, const float* __restrict__ q,
    const float* __restrict__ prel, float* __restrict__ alpha,
    unsigned* __restrict__ amax, int E) {
  int i = blockIdx.x * 256 + threadIdx.x;
  if (i >= E * 8) return;
  int e = i >> 3, h = i & 7;
  int s = src[e], d = dst[e];
  const float* kr = kt + (size_t)s * kHID + h * 16;
  const float* qr = q  + (size_t)d * kHID + h * 16;
  float acc = 0.f;
#pragma unroll
  for (int u = 0; u < 16; ++u) acc = fmaf(qr[u], kr[u], acc);
  acc *= prel[h] * kInvSqrtD;
  alpha[i] = acc;
  atomicMax(&amax[(size_t)d * 8 + h], f2mono(acc));   // segment max
}

__global__ __launch_bounds__(256) void k_edge_exp(
    const int* __restrict__ dst, float* __restrict__ alpha,
    const unsigned* __restrict__ amax, float* __restrict__ den, int E) {
  int i = blockIdx.x * 256 + threadIdx.x;
  if (i >= E * 8) return;
  int e = i >> 3, h = i & 7;
  int d = dst[e];
  float am = mono2f(amax[(size_t)d * 8 + h]);
  if (!(am > -3.0e38f && am < 3.0e38f)) am = 0.f;     // isfinite guard
  float ex = expf(alpha[i] - am);
  alpha[i] = ex;
  atomicAdd(&den[(size_t)d * 8 + h], ex);             // segment sum
}

__global__ __launch_bounds__(256) void k_edge_scatter(
    const int* __restrict__ src, const int* __restrict__ dst,
    const float* __restrict__ alpha, const float* __restrict__ den,
    const float* __restrict__ vt, float* __restrict__ out, int E) {
  long long i = (long long)blockIdx.x * 256 + threadIdx.x;
  if (i >= (long long)E * kHID) return;
  int e = (int)(i >> 7), c = (int)(i & 127), h = c >> 4;
  int s = src[e], d = dst[e];
  float w = alpha[(size_t)e * 8 + h] / (den[(size_t)d * 8 + h] + 1e-16f);
  atomicAdd(&out[(size_t)d * kHID + c], vt[(size_t)s * kHID + c] * w);
}

// ------------------------------------------------ combine / gated residual
__device__ __forceinline__ float gelu_exact(float x) {
  return 0.5f * x * (1.f + erff(x * 0.70710678118654752f));
}
__global__ __launch_bounds__(256) void k_min_gelu(
    const float* __restrict__ a, const float* __restrict__ b,
    float* __restrict__ g, int n) {
  int i = blockIdx.x * 256 + threadIdx.x;
  if (i >= n) return;
  float v = a[i];
  if (b) v = fminf(v, b[i]);
  g[i] = gelu_exact(v);
}
__global__ __launch_bounds__(256) void k_blend(
    float* __restrict__ x, const float* __restrict__ o,
    const float* __restrict__ skip, int n) {
  int i = blockIdx.x * 256 + threadIdx.x;
  if (i >= n) return;
  float a = 1.f / (1.f + expf(-skip[0]));
  x[i] = a * o[i] + (1.f - a) * x[i];
}

// ----------------------------------------------------------- readout head
__global__ __launch_bounds__(256) void k_gather(
    const float* __restrict__ xb, const int* __restrict__ node_idx,
    float* __restrict__ h, int n) {
  int i = blockIdx.x * 256 + threadIdx.x;
  if (i >= n) return;
  int row = i >> 7, c = i & 127;
  int g = row >> 2, j = row & 3;
  int idx = g * kPerGraph + node_idx[j];
  h[i] = xb[(size_t)idx * kHID + c];
}
__global__ __launch_bounds__(256) void k_final(
    const float* __restrict__ h, const float* __restrict__ W3,
    const float* __restrict__ b3, float* __restrict__ out, int n) {
  int i = blockIdx.x * 256 + threadIdx.x;
  if (i >= n * 4) return;
  int row = i >> 2, o = i & 3;
  const float* hr = h + (size_t)row * kHID;
  float acc = b3[o];
#pragma unroll 16
  for (int k = 0; k < kHID; ++k) acc = fmaf(hr[k], W3[k * 4 + o], acc);
  out[i] = acc;
}

// ================================================================= driver
extern "C" void kernel_launch(void* const* d_in, const int* in_sizes, int n_in,
                              void* d_out, int out_size, void* d_ws, size_t ws_size,
                              hipStream_t stream) {
  (void)in_sizes; (void)n_in; (void)out_size; (void)ws_size;
  const float* x_bus  = (const float*)d_in[0];
  const float* x_gen  = (const float*)d_in[1];
  const float* Wi_bus = (const float*)d_in[2];
  const float* bi_bus = (const float*)d_in[3];
  const float* Wi_gen = (const float*)d_in[4];
  const float* bi_gen = (const float*)d_in[5];
  const float* Wk = (const float*)d_in[6];
  const float* bk = (const float*)d_in[7];
  const float* Wq = (const float*)d_in[8];
  const float* bq = (const float*)d_in[9];
  const float* Wv = (const float*)d_in[10];
  const float* bv = (const float*)d_in[11];
  const float* Wa = (const float*)d_in[12];
  const float* ba = (const float*)d_in[13];
  const float* skip  = (const float*)d_in[14];
  const float* a_rel = (const float*)d_in[15];
  const float* m_rel = (const float*)d_in[16];
  const float* p_rel = (const float*)d_in[17];
  const float* W1 = (const float*)d_in[18];
  const float* b1 = (const float*)d_in[19];
  const float* W2 = (const float*)d_in[20];
  const float* b2 = (const float*)d_in[21];
  const float* W3 = (const float*)d_in[22];
  const float* b3 = (const float*)d_in[23];
  const int* ei_bb = (const int*)d_in[24];
  const int* ei_gb = (const int*)d_in[25];
  const int* ei_bg = (const int*)d_in[26];
  const int* node_idx = (const int*)d_in[27];
  // d_in[28] = num_graphs (known constant 10000)

  // ---- workspace carve (f32 elements), ~1.0 GB total
  float* ws = (float*)d_ws;
  size_t off = 0;
  auto take = [&](size_t n) { float* p = ws + off; off += n; return p; };
  float* xb  = take((size_t)kNBp * kHID);
  float* xg  = take((size_t)kNGp * kHID);
  float* kb  = take((size_t)kNBp * kHID);
  float* qb  = take((size_t)kNBp * kHID);
  float* vb  = take((size_t)kNBp * kHID);
  float* kg  = take((size_t)kNGp * kHID);
  float* qg  = take((size_t)kNGp * kHID);
  float* vg  = take((size_t)kNGp * kHID);
  float* kt  = take((size_t)kNBp * kHID);   // relation-transformed K (reused)
  float* vt  = take((size_t)kNBp * kHID);   // relation-transformed V (reused)
  float* obb = take((size_t)kNBp * kHID);
  float* ogb = take((size_t)kNBp * kHID);
  float* obg = take((size_t)kNGp * kHID);
  float* alpha = take((size_t)kEBB * 8);
  unsigned* amax = (unsigned*)take((size_t)kNBp * 8);
  float* den = take((size_t)kNBp * 8);
  float* WblkA = take(16384);
  float* WblkM = take(16384);
  float* h0 = take((size_t)kNH * kHID);
  float* h1 = take((size_t)kNH * kHID);

  const int TPB = 256;
  auto blocks = [](long long n) { return (unsigned)((n + 255) / 256); };
  const int gBus = kNBp / 64, gGen = kNGp / 64, gH = kNH / 64;

  k_in_linear_relu<<<blocks((long long)kNB * kHID), TPB, 0, stream>>>(x_bus, Wi_bus, bi_bus, xb, kNB);
  k_in_linear_relu<<<blocks((long long)kNG * kHID), TPB, 0, stream>>>(x_gen, Wi_gen, bi_gen, xg, kNG);

  for (int l = 0; l < 2; ++l) {
    // ---- K/Q/V projections (WMMA f16, f32 accumulate)
    const float* xs[2] = {xb, xg};
    float* ks[2] = {kb, kg};
    float* qs[2] = {qb, qg};
    float* vs[2] = {vb, vg};
    const int gN[2] = {gBus, gGen};
    for (int ty = 0; ty < 2; ++ty) {
      size_t wo = (size_t)(l * 2 + ty) * kHID * kHID;
      size_t bo = (size_t)(l * 2 + ty) * kHID;
      k_gemm128_wmma<<<gN[ty], TPB, 0, stream>>>(xs[ty], Wk + wo, bk + bo, ks[ty], 0);
      k_gemm128_wmma<<<gN[ty], TPB, 0, stream>>>(xs[ty], Wq + wo, bq + bo, qs[ty], 0);
      k_gemm128_wmma<<<gN[ty], TPB, 0, stream>>>(xs[ty], Wv + wo, bv + bo, vs[ty], 0);
    }

    // ---- three HGT relations
    struct Rel { const float *k, *v, *q; const int* ei; int E, gSrc; size_t ndstp; float* out; };
    Rel rels[3] = {
      {kb, vb, qb, ei_bb, kEBB, gBus, (size_t)kNBp, obb},   // bus->bus
      {kg, vg, qb, ei_gb, kEGB, gGen, (size_t)kNBp, ogb},   // gen->bus
      {kb, vb, qg, ei_bg, kEBG, gBus, (size_t)kNGp, obg},   // bus->gen
    };
    for (int r = 0; r < 3; ++r) {
      const Rel& R = rels[r];
      size_t ro = (size_t)(l * 3 + r) * 2048;   // H*D*D = 8*16*16
      k_blockdiag<<<64, TPB, 0, stream>>>(a_rel + ro, WblkA);
      k_blockdiag<<<64, TPB, 0, stream>>>(m_rel + ro, WblkM);
      // per-head 16x16 transforms as one block-diagonal 128x128 WMMA GEMM
      k_gemm128_wmma<<<R.gSrc, TPB, 0, stream>>>(R.k, WblkA, nullptr, kt, 0);
      k_gemm128_wmma<<<R.gSrc, TPB, 0, stream>>>(R.v, WblkM, nullptr, vt, 0);
      hipMemsetAsync(amax, 0, R.ndstp * 8 * sizeof(unsigned), stream);
      hipMemsetAsync(den, 0, R.ndstp * 8 * sizeof(float), stream);
      hipMemsetAsync(R.out, 0, R.ndstp * kHID * sizeof(float), stream);
      const int* src = R.ei;
      const int* dst = R.ei + R.E;
      const float* pr = p_rel + (size_t)(l * 3 + r) * 8;
      k_edge_alpha<<<blocks((long long)R.E * 8), TPB, 0, stream>>>(src, dst, kt, R.q, pr, alpha, amax, R.E);
      k_edge_exp<<<blocks((long long)R.E * 8), TPB, 0, stream>>>(dst, alpha, amax, den, R.E);
      k_edge_scatter<<<blocks((long long)R.E * kHID), TPB, 0, stream>>>(src, dst, alpha, den, vt, R.out, R.E);
    }

    // ---- bus: min-combine, gelu, out-proj, gated skip (kt/vt reused as scratch)
    {
      int n = kNBp * kHID;
      k_min_gelu<<<blocks(n), TPB, 0, stream>>>(obb, ogb, kt, n);
      size_t wo = (size_t)(l * 2 + 0) * kHID * kHID, bo = (size_t)(l * 2 + 0) * kHID;
      k_gemm128_wmma<<<gBus, TPB, 0, stream>>>(kt, Wa + wo, ba + bo, vt, 0);
      k_blend<<<blocks(n), TPB, 0, stream>>>(xb, vt, skip + (size_t)l * 2 + 0, n);
    }
    // ---- gen
    {
      int n = kNGp * kHID;
      k_min_gelu<<<blocks(n), TPB, 0, stream>>>(obg, nullptr, kt, n);
      size_t wo = (size_t)(l * 2 + 1) * kHID * kHID, bo = (size_t)(l * 2 + 1) * kHID;
      k_gemm128_wmma<<<gGen, TPB, 0, stream>>>(kt, Wa + wo, ba + bo, vt, 0);
      k_blend<<<blocks(n), TPB, 0, stream>>>(xg, vt, skip + (size_t)l * 2 + 1, n);
    }
  }

  // ---- readout: gather 4 nodes/graph then 2x relu-MLP (WMMA) + final 128->4
  k_gather<<<blocks((long long)kNH * kHID), TPB, 0, stream>>>(xb, node_idx, h0, kNH * kHID);
  k_gemm128_wmma<<<gH, TPB, 0, stream>>>(h0, W1, b1, h1, 1);
  k_gemm128_wmma<<<gH, TPB, 0, stream>>>(h1, W2, b2, h0, 1);
  k_final<<<blocks((long long)kNH * 4), TPB, 0, stream>>>(h0, W3, b3, (float*)d_out, kNH);
}
